// BaselineGNN_35897336660281
// MI455X (gfx1250) — matrix operand order repaired and checked by reading, not compiled, not run on hardware
//
#include <hip/hip_runtime.h>
#include <math.h>

#define NN     50000
#define NEDGE  200000
#define NGRAPH 64
#define NHEAD  8
#define HDIM   48
#define FDIM   384
#define LN_EPS 1e-5f

typedef __attribute__((ext_vector_type(16))) _Float16 v16h;
typedef __attribute__((ext_vector_type(8)))  _Float16 v8h;
typedef __attribute__((ext_vector_type(8)))  float    v8f;

// ---------------------------------------------------------------------------
// WMMA GEMM, 16x64 strip per wave, register double-buffered K pipeline.
//   C[M,Ncols] = A[M,K] (f16 row-major, K mult of 32) @ Bpacked + bias
// B packed per (kt,nt) 32x16 tile, lane-major (512 halves/tile):
//   lane<16 : half i -> B[kt*32+(i&7)+(i>>3)*16  ][nt*16+lane]
//   lane>=16: half i -> B[kt*32+8+(i&7)+(i>>3)*16][nt*16+lane-16]
// A fragment (ISA 16-bit A 16x32): lane L row M=L%16,
//   halves 0..7  -> K = base..base+7   (base = L<16 ? 0 : 8)
//   halves 8..15 -> K = base+16..base+23
// C/D: VGPR r, lane l: row = r + (l<16?0:8), col = l&15.
// Block = 4 waves; wave w handles M-tile blockIdx.x*4+w. Ntp must be mult of 4.
// ---------------------------------------------------------------------------
__global__ void k_gemm(const _Float16* __restrict__ A, const _Float16* __restrict__ Bp,
                       const float* __restrict__ bias,
                       float* __restrict__ C, int ldc,
                       _Float16* __restrict__ Ch, int ldch,
                       int M, int K, int Ncols, int Ntp, int relu)
{
    const int tm   = blockIdx.x * 4 + threadIdx.y;
    const int tn0  = blockIdx.y * 4;            // 4 consecutive n-tiles
    const int lane = threadIdx.x;

    int row = tm * 16 + (lane & 15);
    if (row >= M) row = M - 1;                  // clamp loads, guard stores later
    const int k0 = (lane < 16) ? 0 : 8;

    const _Float16* ap = A + (size_t)row * K + k0;
    const _Float16* bp = Bp + ((size_t)((tn0 * 32) + lane) << 4);
    const size_t bstep = (size_t)Ntp << 9;      // Ntp*512 halves per K-tile row

    v8f acc[4] = {{}, {}, {}, {}};
    const int Kt = K >> 5;

    // prologue: load fragments for kt = 0
    v8h  lo = *(const v8h*)(ap);
    v8h  hi = *(const v8h*)(ap + 16);
    v16h b0 = *(const v16h*)(bp);
    v16h b1 = *(const v16h*)(bp + 512);
    v16h b2 = *(const v16h*)(bp + 1024);
    v16h b3 = *(const v16h*)(bp + 1536);

    for (int kt = 0; kt < Kt - 1; ++kt) {
        ap += 32;
        bp += bstep;
        __builtin_prefetch(ap + 512, 0, 0);     // speculative A-stream prefetch
        // double buffer: fetch next K-step while WMMAs consume current
        v8h  nlo = *(const v8h*)(ap);
        v8h  nhi = *(const v8h*)(ap + 16);
        v16h nb0 = *(const v16h*)(bp);
        v16h nb1 = *(const v16h*)(bp + 512);
        v16h nb2 = *(const v16h*)(bp + 1024);
        v16h nb3 = *(const v16h*)(bp + 1536);

        v16h a;
#pragma unroll
        for (int i = 0; i < 8; ++i) { a[i] = lo[i]; a[i + 8] = hi[i]; }
        acc[0] = __builtin_amdgcn_wmma_f32_16x16x32_f16(false, a, false, b0, (short)0, acc[0], false, false);
        acc[1] = __builtin_amdgcn_wmma_f32_16x16x32_f16(false, a, false, b1, (short)0, acc[1], false, false);
        acc[2] = __builtin_amdgcn_wmma_f32_16x16x32_f16(false, a, false, b2, (short)0, acc[2], false, false);
        acc[3] = __builtin_amdgcn_wmma_f32_16x16x32_f16(false, a, false, b3, (short)0, acc[3], false, false);

        lo = nlo; hi = nhi; b0 = nb0; b1 = nb1; b2 = nb2; b3 = nb3;
    }
    {   // epilogue K-step
        v16h a;
#pragma unroll
        for (int i = 0; i < 8; ++i) { a[i] = lo[i]; a[i + 8] = hi[i]; }
        acc[0] = __builtin_amdgcn_wmma_f32_16x16x32_f16(false, a, false, b0, (short)0, acc[0], false, false);
        acc[1] = __builtin_amdgcn_wmma_f32_16x16x32_f16(false, a, false, b1, (short)0, acc[1], false, false);
        acc[2] = __builtin_amdgcn_wmma_f32_16x16x32_f16(false, a, false, b2, (short)0, acc[2], false, false);
        acc[3] = __builtin_amdgcn_wmma_f32_16x16x32_f16(false, a, false, b3, (short)0, acc[3], false, false);
    }

    const int rbase = tm * 16 + ((lane < 16) ? 0 : 8);
#pragma unroll
    for (int j = 0; j < 4; ++j) {
        const int col = (tn0 + j) * 16 + (lane & 15);
        const float bv = (bias && col < Ncols) ? bias[col] : 0.0f;
#pragma unroll
        for (int r = 0; r < 8; ++r) {
            const int rr = rbase + r;
            float val = acc[j][r] + bv;
            if (relu) val = fmaxf(val, 0.0f);
            if (rr < M && col < Ncols) {
                if (C)  C[(size_t)rr * ldc + col]  = val;
                if (Ch) Ch[(size_t)rr * ldch + col] = (_Float16)val;
            }
        }
    }
}

// Pack f32 weight [K,Ncols] (row-major) into WMMA B-fragment f16 layout.
__global__ void k_pack(const float* __restrict__ W, _Float16* __restrict__ P,
                       int K, int Ncols, int Ntp, int total)
{
    int idx = blockIdx.x * blockDim.x + threadIdx.x;
    if (idx >= total) return;
    const int half_i = idx & 15;
    const int lane   = (idx >> 4) & 31;
    const int tile   = idx >> 9;
    const int nt     = tile % Ntp;
    const int kt     = tile / Ntp;
    const int k = kt * 32 + ((lane < 16) ? 0 : 8) + (half_i & 7) + ((half_i >> 3) << 4);
    const int n = nt * 16 + (lane & 15);
    float v = (k < K && n < Ncols) ? W[(size_t)k * Ncols + n] : 0.0f;
    P[idx] = (_Float16)v;
}

// f32 [rows,K] -> f16 [rows,Kp], zero padded (Kp multiple of 32).
__global__ void k_cvt_pad(const float* __restrict__ src, _Float16* __restrict__ dst,
                          size_t rows, int K, int Kp)
{
    size_t i = blockIdx.x * (size_t)blockDim.x + threadIdx.x;
    if (i >= rows * (size_t)Kp) return;
    size_t r = i / Kp;
    int c = (int)(i % Kp);
    dst[i] = (c < K) ? (_Float16)src[r * K + c] : (_Float16)0.0f;
}

__global__ void k_fill(float* p, float v, size_t n)
{
    size_t i = blockIdx.x * (size_t)blockDim.x + threadIdx.x;
    if (i < n) p[i] = v;
}

// sum / sumsq reduction -> stats[0], stats[1]
__global__ void k_reduce(const float* __restrict__ x, size_t n, float* stats)
{
    __shared__ float s0[256], s1[256];
    size_t i = blockIdx.x * (size_t)blockDim.x + threadIdx.x;
    float a = 0.f, b = 0.f;
    for (size_t idx = i; idx < n; idx += (size_t)gridDim.x * blockDim.x) {
        float v = x[idx]; a += v; b += v * v;
    }
    s0[threadIdx.x] = a; s1[threadIdx.x] = b;
    __syncthreads();
    for (int o = 128; o > 0; o >>= 1) {
        if ((int)threadIdx.x < o) {
            s0[threadIdx.x] += s0[threadIdx.x + o];
            s1[threadIdx.x] += s1[threadIdx.x + o];
        }
        __syncthreads();
    }
    if (threadIdx.x == 0) { atomicAdd(&stats[0], s0[0]); atomicAdd(&stats[1], s1[0]); }
}

// global LayerNorm apply: (x - mean)*rstd*g + b, optional f32/f16 outputs
__global__ void k_ln_apply(const float* __restrict__ x,
                           const float* __restrict__ gamma, const float* __restrict__ beta,
                           const float* __restrict__ stats, float invn,
                           size_t rows, int cols,
                           float* __restrict__ outF, int ldf,
                           _Float16* __restrict__ outH, int ldh)
{
    size_t i = blockIdx.x * (size_t)blockDim.x + threadIdx.x;
    if (i >= rows * (size_t)cols) return;
    size_t r = i / cols;
    int c = (int)(i % cols);
    float mean = stats[0] * invn;
    float var  = stats[1] * invn - mean * mean;
    float rstd = rsqrtf(var + LN_EPS);
    float v = (x[i] - mean) * rstd * gamma[c] + beta[c];
    if (outF) outF[r * (size_t)ldf + c] = v;
    if (outH) outH[r * (size_t)ldh + c] = (_Float16)v;
}

__device__ __forceinline__ void atomicMaxFloat(float* addr, float val)
{
    unsigned int* ua = (unsigned int*)addr;
    unsigned int old = *ua;
    while (__uint_as_float(old) < val) {
        unsigned int assumed = old;
        old = atomicCAS(ua, assumed, __float_as_uint(val));
        if (old == assumed) break;
    }
}

// alpha[e,h] = <q[dst], k[src]+ee[e]> / sqrt(D); segment max over dst
__global__ void k_edge_alpha(const float* __restrict__ q, const float* __restrict__ k,
                             const _Float16* __restrict__ ee,
                             const int* __restrict__ src, const int* __restrict__ dst,
                             float* __restrict__ alpha, float* __restrict__ mmax)
{
    int t = blockIdx.x * blockDim.x + threadIdx.x;
    if (t >= NEDGE * NHEAD) return;
    int e = t >> 3, h = t & 7;
    int s = src[e], d = dst[e];
    const float*    qp = q  + (size_t)d * FDIM + h * HDIM;
    const float*    kp = k  + (size_t)s * FDIM + h * HDIM;
    const _Float16* ep = ee + (size_t)e * FDIM + h * HDIM;
    float acc = 0.f;
#pragma unroll 8
    for (int i = 0; i < HDIM; ++i) acc += qp[i] * (kp[i] + (float)ep[i]);
    acc *= 0.14433756729740643f;   // 1/sqrt(48)
    alpha[t] = acc;
    atomicMaxFloat(&mmax[(size_t)d * NHEAD + h], acc);
}

// ex = exp(alpha - m[dst]); segment sum -> ssum (alpha overwritten with ex)
__global__ void k_edge_exp(float* __restrict__ alpha, const float* __restrict__ mmax,
                           const int* __restrict__ dst, float* __restrict__ ssum)
{
    int t = blockIdx.x * blockDim.x + threadIdx.x;
    if (t >= NEDGE * NHEAD) return;
    int e = t >> 3, h = t & 7;
    int d = dst[e];
    float ex = expf(alpha[t] - mmax[(size_t)d * NHEAD + h]);
    alpha[t] = ex;
    atomicAdd(&ssum[(size_t)d * NHEAD + h], ex);
}

// out[dst] += (v[src] + ee) * a ; one thread per (edge, head, 4-channel chunk)
// -> 12x more threads, only 4 dependent atomics per thread
#define MSG_CHUNKS 12                       /* 48 / 4 */
__global__ void k_edge_msg(const float* __restrict__ ex, const float* __restrict__ ssum,
                           const float* __restrict__ v, const _Float16* __restrict__ ee,
                           const int* __restrict__ src, const int* __restrict__ dst,
                           float* __restrict__ out)
{
    int t = blockIdx.x * blockDim.x + threadIdx.x;
    if (t >= NEDGE * NHEAD * MSG_CHUNKS) return;
    int e  = t / (NHEAD * MSG_CHUNKS);
    int j  = t % (NHEAD * MSG_CHUNKS);
    int h  = j / MSG_CHUNKS;
    int c0 = (j % MSG_CHUNKS) * 4;
    int s = src[e], d = dst[e];
    float a = ex[(size_t)e * NHEAD + h] / (ssum[(size_t)d * NHEAD + h] + 1e-16f);
    const float*    vp = v   + (size_t)s * FDIM + h * HDIM + c0;
    const _Float16* ep = ee  + (size_t)e * FDIM + h * HDIM + c0;
    float*          op = out + (size_t)d * FDIM + h * HDIM + c0;
#pragma unroll
    for (int i = 0; i < 4; ++i)
        atomicAdd(&op[i], (vp[i] + (float)ep[i]) * a);
}

// out = h + elu(out)    (in place on out)
__global__ void k_res_elu(const float* __restrict__ h, float* __restrict__ out, size_t n)
{
    size_t i = blockIdx.x * (size_t)blockDim.x + threadIdx.x;
    if (i >= n) return;
    float t = out[i];
    float e = (t > 0.f) ? t : expm1f(t);
    out[i] = h[i] + e;
}

// per-graph stats: one thread per node, accumulate sum/sumsq/count*C
__global__ void k_gstats(const float* __restrict__ h, const int* __restrict__ batch,
                         float* __restrict__ gsum, float* __restrict__ gsq,
                         float* __restrict__ gcnt)
{
    int i = blockIdx.x * blockDim.x + threadIdx.x;
    if (i >= NN) return;
    int g = batch[i];
    const float* p = h + (size_t)i * FDIM;
    float a = 0.f, b = 0.f;
#pragma unroll 8
    for (int c = 0; c < FDIM; ++c) { float v = p[c]; a += v; b += v * v; }
    atomicAdd(&gsum[g], a);
    atomicAdd(&gsq[g],  b);
    atomicAdd(&gcnt[g], (float)FDIM);
}

__global__ void k_gapply(const float* __restrict__ x, const int* __restrict__ batch,
                         const float* __restrict__ gsum, const float* __restrict__ gsq,
                         const float* __restrict__ gcnt,
                         const float* __restrict__ gamma, const float* __restrict__ beta,
                         float* __restrict__ outF, _Float16* __restrict__ outH)
{
    size_t i = blockIdx.x * (size_t)blockDim.x + threadIdx.x;
    if (i >= (size_t)NN * FDIM) return;
    int node = (int)(i / FDIM);
    int c    = (int)(i % FDIM);
    int g    = batch[node];
    float cnt  = fmaxf(gcnt[g], 1.0f);
    float mean = gsum[g] / cnt;
    float var  = gsq[g] / cnt - mean * mean;
    float rstd = rsqrtf(var + LN_EPS);
    float v = (x[i] - mean) * rstd * gamma[c] + beta[c];
    outF[i] = v;
    outH[i] = (_Float16)v;
}

// ---------------------------------------------------------------------------
// Host side
// ---------------------------------------------------------------------------
static inline void launch_gemm(hipStream_t st, const _Float16* A, const _Float16* Bp,
                               const float* bias, float* C, int ldc,
                               _Float16* Ch, int ldch,
                               int M, int Kp, int Ncols, int relu)
{
    int Mt  = (M + 15) / 16;
    int Ntp = (Ncols + 15) / 16;        // always a multiple of 4 in this model
    dim3 g((Mt + 3) / 4, Ntp / 4);
    dim3 b(32, 4);
    k_gemm<<<g, b, 0, st>>>(A, Bp, bias, C, ldc, Ch, ldch, M, Kp, Ncols, Ntp, relu);
}

static inline void launch_pack(hipStream_t st, const float* W, _Float16* P, int K, int Ncols)
{
    int Kt = (K + 31) / 32, Ntp = (Ncols + 15) / 16;
    int total = Kt * Ntp * 512;
    k_pack<<<(total + 255) / 256, 256, 0, st>>>(W, P, K, Ncols, Ntp, total);
}

static inline size_t packed_halves(int K, int Ncols)
{
    return (size_t)((K + 31) / 32) * ((Ncols + 15) / 16) * 512;
}

extern "C" void kernel_launch(void* const* d_in, const int* in_sizes, int n_in,
                              void* d_out, int out_size, void* d_ws, size_t ws_size,
                              hipStream_t stream)
{
    (void)in_sizes; (void)n_in; (void)out_size; (void)ws_size;

    const float* x_vis  = (const float*)d_in[0];
    const float* x_geo  = (const float*)d_in[1];
    const float* x_pri  = (const float*)d_in[2];
    const float* e_attr = (const float*)d_in[3];
    const int*   ei     = (const int*)d_in[4];
    const int*   batch  = (const int*)d_in[5];
    const int* src = ei;
    const int* dst = ei + NEDGE;

    // bump allocator over workspace (256B aligned)
    char* wptr = (char*)d_ws;
    auto alloc = [&](size_t bytes) -> void* {
        void* p = (void*)wptr;
        wptr += (bytes + 255) & ~(size_t)255;
        return p;
    };

    // f16 activation inputs
    _Float16* xv_h = (_Float16*)alloc((size_t)NN * 1024 * 2);
    _Float16* xg_h = (_Float16*)alloc((size_t)NN * 32 * 2);
    _Float16* xp_h = (_Float16*)alloc((size_t)NN * 64 * 2);
    _Float16* ea_h = (_Float16*)alloc((size_t)NEDGE * 32 * 2);
    // features
    float*    h      = (float*)alloc((size_t)NN * FDIM * 4);
    _Float16* h_half = (_Float16*)alloc((size_t)NN * FDIM * 2);
    _Float16* ee_enc_h = (_Float16*)alloc((size_t)NEDGE * 128 * 2);   // encoded edge attrs (f16)
    float*    tmp    = (float*)alloc((size_t)NEDGE * 128 * 4);        // max(N,E) x 128
    _Float16* tmp_h  = (_Float16*)alloc((size_t)NN * 128 * 2);
    float*    q      = (float*)alloc((size_t)NN * FDIM * 4);
    float*    kbuf   = (float*)alloc((size_t)NN * FDIM * 4);
    float*    v      = (float*)alloc((size_t)NN * FDIM * 4);
    float*    outb   = (float*)alloc((size_t)NN * FDIM * 4);
    _Float16* ee     = (_Float16*)alloc((size_t)NEDGE * FDIM * 2);    // per-layer edge proj
    float*    alpha  = (float*)alloc((size_t)NEDGE * NHEAD * 4);
    float*    mmax   = (float*)alloc((size_t)NN * NHEAD * 4);
    float*    ssum   = (float*)alloc((size_t)NN * NHEAD * 4);
    float*    stats  = (float*)alloc(2 * 4);
    float*    gsum   = (float*)alloc(NGRAPH * 4);
    float*    gsq    = (float*)alloc(NGRAPH * 4);
    float*    gcnt   = (float*)alloc(NGRAPH * 4);

    // packed weights
    _Float16* p_vis = (_Float16*)alloc(packed_halves(1024, 128) * 2);
    _Float16* p_geo = (_Float16*)alloc(packed_halves(6, 128) * 2);
    _Float16* p_pri = (_Float16*)alloc(packed_halves(50, 128) * 2);
    _Float16* p_edg = (_Float16*)alloc(packed_halves(3, 128) * 2);
    _Float16* p_q[2], *p_k[2], *p_v[2], *p_s[2], *p_e[2];
    for (int l = 0; l < 2; ++l) {
        p_q[l] = (_Float16*)alloc(packed_halves(384, 384) * 2);
        p_k[l] = (_Float16*)alloc(packed_halves(384, 384) * 2);
        p_v[l] = (_Float16*)alloc(packed_halves(384, 384) * 2);
        p_s[l] = (_Float16*)alloc(packed_halves(384, 384) * 2);
        p_e[l] = (_Float16*)alloc(packed_halves(128, 384) * 2);
    }
    _Float16* p_c1 = (_Float16*)alloc(packed_halves(384, 128) * 2);
    _Float16* p_c2 = (_Float16*)alloc(packed_halves(128, 49) * 2);

    // layer weight input indices: Wq,bq,Wk,bk,Wv,bv,We,Ws,bs,g,bt
    const int LI[2][11] = {{22,23,24,25,26,27,28,29,30,31,32},
                           {33,34,35,36,37,38,39,40,41,42,43}};

    // ---- pack all weights ----
    launch_pack(stream, (const float*)d_in[6],  p_vis, 1024, 128);
    launch_pack(stream, (const float*)d_in[10], p_geo, 6,    128);
    launch_pack(stream, (const float*)d_in[14], p_pri, 50,   128);
    launch_pack(stream, (const float*)d_in[18], p_edg, 3,    128);
    for (int l = 0; l < 2; ++l) {
        launch_pack(stream, (const float*)d_in[LI[l][0]], p_q[l], 384, 384);
        launch_pack(stream, (const float*)d_in[LI[l][2]], p_k[l], 384, 384);
        launch_pack(stream, (const float*)d_in[LI[l][4]], p_v[l], 384, 384);
        launch_pack(stream, (const float*)d_in[LI[l][7]], p_s[l], 384, 384);
        launch_pack(stream, (const float*)d_in[LI[l][6]], p_e[l], 128, 384);
    }
    launch_pack(stream, (const float*)d_in[44], p_c1, 384, 128);
    launch_pack(stream, (const float*)d_in[46], p_c2, 128, 49);

    // ---- convert inputs to f16 (K padded to mult of 32) ----
    auto cvt = [&](const float* s, _Float16* dh, size_t rows, int K, int Kp) {
        size_t n = rows * (size_t)Kp;
        k_cvt_pad<<<(unsigned)((n + 255) / 256), 256, 0, stream>>>(s, dh, rows, K, Kp);
    };
    cvt(x_vis,  xv_h, NN, 1024, 1024);
    cvt(x_geo,  xg_h, NN, 6,  32);
    cvt(x_pri,  xp_h, NN, 50, 64);
    cvt(e_attr, ea_h, NEDGE, 3, 32);

    // ---- encoders: Linear+ReLU -> global LayerNorm -> column slice of h ----
    struct Enc { const _Float16* A; const _Float16* P; int Kp; int bi, gi, bei; int coloff; };
    Enc encs[3] = {{xv_h, p_vis, 1024, 7,  8,  9,  0},
                   {xg_h, p_geo, 32,   11, 12, 13, 128},
                   {xp_h, p_pri, 64,   15, 16, 17, 256}};
    for (int ec = 0; ec < 3; ++ec) {
        launch_gemm(stream, encs[ec].A, encs[ec].P, (const float*)d_in[encs[ec].bi],
                    tmp, 128, nullptr, 0, NN, encs[ec].Kp, 128, 1);
        k_fill<<<1, 2, 0, stream>>>(stats, 0.f, 2);
        size_t n = (size_t)NN * 128;
        k_reduce<<<1024, 256, 0, stream>>>(tmp, n, stats);
        k_ln_apply<<<(unsigned)((n + 255) / 256), 256, 0, stream>>>(
            tmp, (const float*)d_in[encs[ec].gi], (const float*)d_in[encs[ec].bei],
            stats, 1.0f / (float)n, NN, 128,
            h + encs[ec].coloff, FDIM, h_half + encs[ec].coloff, FDIM);
    }
    // edge encoder -> ee_enc_h (f16 only)
    {
        launch_gemm(stream, ea_h, p_edg, (const float*)d_in[19],
                    tmp, 128, nullptr, 0, NEDGE, 32, 128, 1);
        k_fill<<<1, 2, 0, stream>>>(stats, 0.f, 2);
        size_t n = (size_t)NEDGE * 128;
        k_reduce<<<1024, 256, 0, stream>>>(tmp, n, stats);
        k_ln_apply<<<(unsigned)((n + 255) / 256), 256, 0, stream>>>(
            tmp, (const float*)d_in[20], (const float*)d_in[21],
            stats, 1.0f / (float)n, NEDGE, 128,
            nullptr, 0, ee_enc_h, 128);
    }

    // ---- two TransformerConv layers ----
    const unsigned EHB  = (NEDGE * NHEAD + 255) / 256;
    const unsigned EMB  = ((unsigned)NEDGE * NHEAD * MSG_CHUNKS + 255) / 256;
    const size_t   NC   = (size_t)NN * FDIM;
    for (int l = 0; l < 2; ++l) {
        // q,k,v and skip (skip initializes outb)
        launch_gemm(stream, h_half, p_q[l], (const float*)d_in[LI[l][1]], q,    FDIM, nullptr, 0, NN, 384, 384, 0);
        launch_gemm(stream, h_half, p_k[l], (const float*)d_in[LI[l][3]], kbuf, FDIM, nullptr, 0, NN, 384, 384, 0);
        launch_gemm(stream, h_half, p_v[l], (const float*)d_in[LI[l][5]], v,    FDIM, nullptr, 0, NN, 384, 384, 0);
        launch_gemm(stream, h_half, p_s[l], (const float*)d_in[LI[l][8]], outb, FDIM, nullptr, 0, NN, 384, 384, 0);
        // edge projection ee = e_enc @ We (no bias), f16 output
        launch_gemm(stream, ee_enc_h, p_e[l], nullptr, nullptr, 0, ee, FDIM, NEDGE, 128, 384, 0);

        // scatter softmax
        k_fill<<<(NN * NHEAD + 255) / 256, 256, 0, stream>>>(mmax, -3.0e38f, (size_t)NN * NHEAD);
        k_fill<<<(NN * NHEAD + 255) / 256, 256, 0, stream>>>(ssum, 0.f,      (size_t)NN * NHEAD);
        k_edge_alpha<<<EHB, 256, 0, stream>>>(q, kbuf, ee, src, dst, alpha, mmax);
        k_edge_exp<<<EHB, 256, 0, stream>>>(alpha, mmax, dst, ssum);
        k_edge_msg<<<EMB, 256, 0, stream>>>(alpha, ssum, v, ee, src, dst, outb);

        // residual + elu, then per-graph LayerNorm back into h / h_half
        k_res_elu<<<(unsigned)((NC + 255) / 256), 256, 0, stream>>>(h, outb, NC);
        k_fill<<<1, 64, 0, stream>>>(gsum, 0.f, NGRAPH);
        k_fill<<<1, 64, 0, stream>>>(gsq,  0.f, NGRAPH);
        k_fill<<<1, 64, 0, stream>>>(gcnt, 0.f, NGRAPH);
        k_gstats<<<(NN + 255) / 256, 256, 0, stream>>>(outb, batch, gsum, gsq, gcnt);
        k_gapply<<<(unsigned)((NC + 255) / 256), 256, 0, stream>>>(
            outb, batch, gsum, gsq, gcnt,
            (const float*)d_in[LI[l][9]], (const float*)d_in[LI[l][10]], h, h_half);
    }

    // ---- classifier ----
    launch_gemm(stream, h_half, p_c1, (const float*)d_in[45], tmp, 128, tmp_h, 128, NN, 384, 128, 1);
    launch_gemm(stream, tmp_h, p_c2, (const float*)d_in[47], (float*)d_out, 49, nullptr, 0, NN, 128, 49, 0);
}
